// RGLRU_25675314495607
// MI455X (gfx1250) — matrix operand order repaired and checked by reading, not compiled, not run on hardware
//
#include <hip/hip_runtime.h>
#include <hip/hip_bf16.h>
#include <math.h>

// ---------------- problem constants (from reference) ----------------
#define BQ   4
#define LQ   2048
#define DQ   4096
#define HQ   8
#define BWQ  512            // D / H
#define MROWS (BQ*LQ)       // 8192 rows for the block-diagonal GEMMs
#define NCHUNK 8            // scan chunks along L
#define CHUNK (LQ/NCHUNK)   // 256

#define ROWS 32             // rows (M) per workgroup = 2 WMMA m-tiles
#define ASTR (BWQ + 8)      // padded LDS row stride (1040 B == 16 B mod 256 B)

typedef _Float16 v16h  __attribute__((ext_vector_type(16)));
typedef _Float16 half8 __attribute__((ext_vector_type(8)));
typedef _Float16 half4 __attribute__((ext_vector_type(4)));
typedef float    v8f   __attribute__((ext_vector_type(8)));

// ---------------------------------------------------------------
// Kernel 0: transpose + f16-convert both weight matrices.
// w[h][i][j] (i = K, j = N) -> wT[h][j][i] so a WMMA B-fragment
// (one output column, contiguous in K) is a contiguous 32B run/lane.
// ---------------------------------------------------------------
__global__ __launch_bounds__(256) void rglru_prep_weights(
    const float* __restrict__ w_in, const float* __restrict__ w_a,
    _Float16* __restrict__ wT_in, _Float16* __restrict__ wT_a)
{
    int t = blockIdx.x * blockDim.x + threadIdx.x;
    if (t >= HQ * BWQ * BWQ) return;
    int h   = t / (BWQ * BWQ);
    int rem = t % (BWQ * BWQ);
    int i   = rem / BWQ;   // K
    int j   = rem % BWQ;   // N
    size_t dst = ((size_t)h * BWQ + j) * BWQ + i;
    wT_in[dst] = (_Float16)w_in[t];
    wT_a[dst]  = (_Float16)w_a[t];
}

// ---------------------------------------------------------------
// Kernel 1: fused dual block-diagonal GEMM (WMMA f16 -> f32) +
// gate/elementwise epilogue. Produces a[] and xn[] for the scan.
//
// Grid: x = 32 slabs (head h = slab>>2, 128-col slab s = slab&3),
//       y = 256 m-blocks (32 rows each).
// Block: 256 threads = 8 waves; wave w owns 16 columns of the slab
// and BOTH 16-row m-tiles (B-fragments reused across the two tiles).
// Whole 32x512 x-tile staged in LDS once (single barrier), then a
// barrier-free K loop: 16 steps x 4 v_wmma_f32_16x16x32_f16.
// ---------------------------------------------------------------
__global__ __launch_bounds__(256) void rglru_gemm_gates(
    const float* __restrict__ x, const int* __restrict__ segpos,
    const float* __restrict__ b_in, const float* __restrict__ b_a,
    const float* __restrict__ a_param,
    const _Float16* __restrict__ wT_in, const _Float16* __restrict__ wT_a,
    float* __restrict__ aArr, float* __restrict__ xnArr)
{
    __shared__ _Float16 As[ROWS * ASTR];   // 32 rows x 512 K (padded), f16

    const int slab = blockIdx.x;           // 0..31
    const int h    = slab >> 2;
    const int s    = slab & 3;
    const int m0   = blockIdx.y * ROWS;    // row block base
    const int tid  = threadIdx.x;
    const int wave = tid >> 5;
    const int lane = tid & 31;
    const int laneLo = lane & 15;
    const int laneHi = lane >> 4;
    const int hbase  = h * BWQ;

    const int jcol = s * 128 + wave * 16 + laneLo;  // column within head [0,512)
    const int d    = hbase + jcol;                  // global feature index

    const _Float16* wcolX = wT_in + ((size_t)h * BWQ + jcol) * BWQ;
    const _Float16* wcolA = wT_a  + ((size_t)h * BWQ + jcol) * BWQ;

    // ---- stage the whole 32x512 x tile as f16 in LDS (one barrier) ----
    // Round q: thread tid handles the float4 at flat index q*1024 + tid*4
    // (perfectly coalesced 16B chunks across the wave).
#pragma unroll
    for (int q = 0; q < (ROWS * BWQ) / (256 * 4); ++q) {
        const int flat = q * 1024 + tid * 4;     // float index in 32x512 tile
        const int r = flat >> 9;                 // /512
        const int c = flat & 511;
        const float4 v = *(const float4*)(x + (size_t)(m0 + r) * DQ + hbase + c);
        half4 hv;
        hv[0] = (_Float16)v.x; hv[1] = (_Float16)v.y;
        hv[2] = (_Float16)v.z; hv[3] = (_Float16)v.w;
        *(half4*)(As + r * ASTR + c) = hv;
    }
    __syncthreads();

    v8f cx0 = {}, ca0 = {};   // m-tile 0 accumulators
    v8f cx1 = {}, ca1 = {};   // m-tile 1 accumulators

#pragma unroll
    for (int k0 = 0; k0 < BWQ; k0 += 32) {
        // A fragments: lane row = laneLo, element e: K = laneHi*16 + e
        // (contiguous, matching the B fragments).
        const _Float16* arow0 = As + laneLo * ASTR + k0 + laneHi * 16;
        const _Float16* arow1 = As + (16 + laneLo) * ASTR + k0 + laneHi * 16;
        half8 a00 = *(const half8*)(arow0);
        half8 a01 = *(const half8*)(arow0 + 8);
        half8 a10 = *(const half8*)(arow1);
        half8 a11 = *(const half8*)(arow1 + 8);
        v16h af0, af1;
#pragma unroll
        for (int e = 0; e < 8; ++e) {
            af0[e] = a00[e]; af0[e + 8] = a01[e];
            af1[e] = a10[e]; af1[e + 8] = a11[e];
        }

        // B fragments: lane holds output column jcol; lanes 0-15 carry
        // K = k0..k0+15, lanes 16-31 carry K = k0+16..k0+31 (32B run).
        v16h bfx = *(const v16h*)(wcolX + k0 + laneHi * 16);
        v16h bfa = *(const v16h*)(wcolA + k0 + laneHi * 16);

        cx0 = __builtin_amdgcn_wmma_f32_16x16x32_f16(
                  false, af0, false, bfx, (short)0, cx0, false, false);
        ca0 = __builtin_amdgcn_wmma_f32_16x16x32_f16(
                  false, af0, false, bfa, (short)0, ca0, false, false);
        cx1 = __builtin_amdgcn_wmma_f32_16x16x32_f16(
                  false, af1, false, bfx, (short)0, cx1, false, false);
        ca1 = __builtin_amdgcn_wmma_f32_16x16x32_f16(
                  false, af1, false, bfa, (short)0, ca1, false, false);
    }

    // ---- fused elementwise epilogue on the C tiles ----
    // C layout: VGPR i -> row M = i + 8*laneHi, col N = laneLo.
    const float bxv = b_in[hbase + jcol];
    const float bav = b_a[hbase + jcol];
    const float ap  = a_param[d];
    const float sp  = (ap > 20.f) ? ap : log1pf(__expf(ap));  // softplus

#pragma unroll
    for (int mt = 0; mt < 2; ++mt) {
        const v8f cx = mt ? cx1 : cx0;
        const v8f ca = mt ? ca1 : ca0;
#pragma unroll
        for (int i = 0; i < 8; ++i) {
            const int r = m0 + mt * 16 + i + 8 * laneHi;   // global row = b*L + l
            const float reset = (segpos[r] == 0) ? 1.f : 0.f;
            const float gx = 1.f / (1.f + __expf(-(cx[i] + bxv)));
            const float ga = 1.f / (1.f + __expf(-(ca[i] + bav)));
            const float log_a = -8.f * ga * sp;
            const float av    = __expf(log_a) * (1.f - reset);
            const float mult  = reset + (1.f - reset) *
                                sqrtf(1.f - __expf(2.f * log_a) + 1e-6f);
            const size_t idx = (size_t)r * DQ + d;
            const float xv = x[idx];
            aArr[idx]  = av;
            xnArr[idx] = xv * gx * mult;
        }
    }
}

// ---------------------------------------------------------------
// Kernel 2 (scan pass 1): per (channel, chunk) compute the chunk's
// decay product P and zero-seeded local final F. 131072 threads.
// ---------------------------------------------------------------
__global__ __launch_bounds__(256) void rglru_scan1(
    const float* __restrict__ aArr, const float* __restrict__ xnArr,
    float* __restrict__ Parr, float* __restrict__ Farr)
{
    const int t = blockIdx.x * blockDim.x + threadIdx.x;
    const int BD = BQ * DQ;
    const int c  = t / BD;            // chunk
    const int ch = t % BD;            // channel (consecutive tids -> consecutive d)
    const int b  = ch / DQ;
    const int dd = ch % DQ;

    size_t idx = ((size_t)b * LQ + (size_t)c * CHUNK) * DQ + dd;
    float y = 0.f, P = 1.f;
#pragma unroll 4
    for (int tt = 0; tt < CHUNK; ++tt) {
        if ((tt & 3) == 0) {   // 16-row-ahead speculative prefetch (OOB-safe)
            __builtin_prefetch(aArr + idx + (size_t)16 * DQ, 0, 1);
            __builtin_prefetch(xnArr + idx + (size_t)16 * DQ, 0, 1);
        }
        const float av = aArr[idx];
        const float xv = xnArr[idx];
        y = fmaf(av, y, xv);
        P *= av;
        idx += DQ;
    }
    Parr[(size_t)c * BD + ch] = P;
    Farr[(size_t)c * BD + ch] = y;
}

// ---------------------------------------------------------------
// Kernel 3 (scan pass 2): compose the 8 affine carries per channel.
// carry[c] = true hidden state entering chunk c. 16384 threads.
// ---------------------------------------------------------------
__global__ __launch_bounds__(256) void rglru_scan2(
    const float* __restrict__ Parr, const float* __restrict__ Farr,
    float* __restrict__ carryArr)
{
    const int ch = blockIdx.x * blockDim.x + threadIdx.x;
    const int BD = BQ * DQ;
    float carry = 0.f;
#pragma unroll
    for (int c = 0; c < NCHUNK; ++c) {
        carryArr[(size_t)c * BD + ch] = carry;
        carry = fmaf(Parr[(size_t)c * BD + ch], carry, Farr[(size_t)c * BD + ch]);
    }
}

// ---------------------------------------------------------------
// Kernel 4 (scan pass 3): re-run each chunk seeded with its carry,
// write y. 131072 threads, coalesced across d.
// ---------------------------------------------------------------
__global__ __launch_bounds__(256) void rglru_scan3(
    const float* __restrict__ aArr, const float* __restrict__ xnArr,
    const float* __restrict__ carryArr, float* __restrict__ out)
{
    const int t = blockIdx.x * blockDim.x + threadIdx.x;
    const int BD = BQ * DQ;
    const int c  = t / BD;
    const int ch = t % BD;
    const int b  = ch / DQ;
    const int dd = ch % DQ;

    float y = carryArr[(size_t)c * BD + ch];
    size_t idx = ((size_t)b * LQ + (size_t)c * CHUNK) * DQ + dd;
#pragma unroll 4
    for (int tt = 0; tt < CHUNK; ++tt) {
        if ((tt & 3) == 0) {
            __builtin_prefetch(aArr + idx + (size_t)16 * DQ, 0, 1);
            __builtin_prefetch(xnArr + idx + (size_t)16 * DQ, 0, 1);
        }
        y = fmaf(aArr[idx], y, xnArr[idx]);
        out[idx] = y;
        idx += DQ;
    }
}

// ---------------------------------------------------------------
extern "C" void kernel_launch(void* const* d_in, const int* in_sizes, int n_in,
                              void* d_out, int out_size, void* d_ws, size_t ws_size,
                              hipStream_t stream) {
    const float* x       = (const float*)d_in[0];
    const int*   segpos  = (const int*)d_in[1];
    // d_in[2] = prev_h : unused by the reference output path
    const float* w_in    = (const float*)d_in[3];
    const float* b_in    = (const float*)d_in[4];
    const float* w_a     = (const float*)d_in[5];
    const float* b_a     = (const float*)d_in[6];
    const float* a_param = (const float*)d_in[7];
    float* out = (float*)d_out;

    // ---- workspace layout ----
    char* ws = (char*)d_ws;
    const size_t WT_BYTES  = (size_t)HQ * BWQ * BWQ * sizeof(_Float16); // 4 MB each
    const size_t BIG_BYTES = (size_t)MROWS * DQ * sizeof(float);        // 134 MB each
    const size_t SM_BYTES  = (size_t)NCHUNK * BQ * DQ * sizeof(float);  // 512 KB each
    _Float16* wT_in = (_Float16*)ws;
    _Float16* wT_a  = (_Float16*)(ws + WT_BYTES);
    float* aArr     = (float*)(ws + 2 * WT_BYTES);
    float* xnArr    = (float*)(ws + 2 * WT_BYTES + BIG_BYTES);
    float* Parr     = (float*)(ws + 2 * WT_BYTES + 2 * BIG_BYTES);
    float* Farr     = (float*)(ws + 2 * WT_BYTES + 2 * BIG_BYTES + SM_BYTES);
    float* carryArr = (float*)(ws + 2 * WT_BYTES + 2 * BIG_BYTES + 2 * SM_BYTES);

    // 0) transpose weights to f16 wT[h][n][k]
    rglru_prep_weights<<<(HQ * BWQ * BWQ) / 256, 256, 0, stream>>>(
        w_in, w_a, wT_in, wT_a);

    // 1) fused dual GEMM (WMMA) + gate epilogue -> a[], xn[]
    rglru_gemm_gates<<<dim3(32, MROWS / ROWS), 256, 0, stream>>>(
        x, segpos, b_in, b_a, a_param, wT_in, wT_a, aArr, xnArr);

    // 2-4) chunked linear-recurrence scan along L
    const int BD = BQ * DQ;
    rglru_scan1<<<(NCHUNK * BD) / 256, 256, 0, stream>>>(aArr, xnArr, Parr, Farr);
    rglru_scan2<<<BD / 256, 256, 0, stream>>>(Parr, Farr, carryArr);
    rglru_scan3<<<(NCHUNK * BD) / 256, 256, 0, stream>>>(aArr, xnArr, carryArr, out);

    (void)in_sizes; (void)n_in; (void)out_size; (void)ws_size;
}
